// EncoderForCoreMLExport_61478161875402
// MI455X (gfx1250) — matrix operand order repaired and checked by reading, not compiled, not run on hardware
//
#include <hip/hip_runtime.h>
#include <cstdint>
#include <cstddef>

// ---------------------------------------------------------------------------
// Types for CDNA5 WMMA (gfx1250, wave32)
// ---------------------------------------------------------------------------
typedef __attribute__((ext_vector_type(16))) __bf16 bf16x16;
typedef __attribute__((ext_vector_type(8)))  float  f32x8;
typedef __attribute__((ext_vector_type(4)))  unsigned int u32x4;
typedef __attribute__((ext_vector_type(8)))  int          i32x8;
typedef __attribute__((ext_vector_type(4)))  int          i32x4;

union FragBF16 {
    bf16x16        v;
    unsigned short u[16];
};

#if defined(__has_builtin)
#  if __has_builtin(__builtin_amdgcn_tensor_load_to_lds) && \
      __has_builtin(__builtin_amdgcn_s_wait_tensorcnt)
#    define HAVE_TDM 1
#  endif
#endif
#ifndef HAVE_TDM
#  define HAVE_TDM 0
#endif

// round-to-nearest-even fp32 -> bf16 (raw bits)
__device__ __forceinline__ unsigned short f2bf(float f) {
    union { float f; unsigned u; } x;
    x.f = f;
    unsigned r = x.u + 0x7FFFu + ((x.u >> 16) & 1u);
    return (unsigned short)(r >> 16);
}

__device__ __forceinline__ float sigf(float x) {
    return 1.0f / (1.0f + __expf(-x));
}
__device__ __forceinline__ float tanh_fast(float x) {
    float e = __expf(2.0f * x);          // 1 - 2/(e^{2x}+1), saturates at +/-1
    return 1.0f - 2.0f / (e + 1.0f);
}

// ---------------------------------------------------------------------------
// fp32 -> bf16 conversion (elementwise, for A-side operands x / outs0)
// ---------------------------------------------------------------------------
__global__ void cvt_f32_bf16(const float* __restrict__ in,
                             unsigned short* __restrict__ out, int n) {
    int i = blockIdx.x * blockDim.x + threadIdx.x;
    if (i < n) out[i] = f2bf(in[i]);
}

// ---------------------------------------------------------------------------
// Repack W [N][K] fp32 row-major into WMMA-B-fragment-contiguous bf16:
//   Wp[((T*(K/32) + Q)*32 + lane)*16 + e]  ==  bf16( W[(T*16 + lane%16)*K
//                                               + Q*32 + (lane/16)*16 + e] )
// so each wave's B fragment for (tile T, k-chunk Q) is one coalesced 1KB block.
// One thread per (T,Q,lane) produces 16 contiguous bf16.
// ---------------------------------------------------------------------------
__global__ void repack_w_bf16(const float* __restrict__ W,
                              unsigned short* __restrict__ Wp,
                              int N, int K) {
    const int id    = blockIdx.x * blockDim.x + threadIdx.x;
    const int Qn    = K >> 5;
    const int total = (N >> 4) * Qn * 32;
    if (id >= total) return;
    const int L  = id & 31;
    const int TQ = id >> 5;
    const int T  = TQ / Qn;
    const int Q  = TQ % Qn;
    const int col   = T * 16 + (L & 15);
    const int kbase = Q * 32 + (L >> 4) * 16;
    unsigned short* dst = &Wp[(size_t)id * 16];
    const float*    src = &W[(size_t)col * K + kbase];
#pragma unroll
    for (int e = 0; e < 16; ++e) dst[e] = f2bf(src[e]);
}

// ---------------------------------------------------------------------------
// C[M,N] = A[M,K](bf16) @ W^T + bias, W given in repacked fragment layout.
// Each wave computes a 16(M) x 64(N) strip: one A fragment feeds 4 WMMAs.
// 256 threads/block = 8 waves. M mult of 16, N mult of 64, K mult of 32.
// ---------------------------------------------------------------------------
__global__ void gemm_bf16_wmma(const unsigned short* __restrict__ A,
                               const unsigned short* __restrict__ Bp,
                               const float* __restrict__ bias,
                               float* __restrict__ C,
                               int M, int N, int K) {
    const int lane = threadIdx.x & 31;
    const int wave = threadIdx.x >> 5;
    const int Nj   = N >> 6;                       // 64-wide jobs along N
    const int job  = blockIdx.x * (blockDim.x >> 5) + wave;
    const int m0   = (job / Nj) << 4;
    const int n0   = (job % Nj) << 6;
    if (m0 >= M) return;

    const int half = lane >> 4;
    const int l16  = lane & 15;
    const int arow = m0 + l16;
    const int Qn   = K >> 5;

    const f32x8 zero = {};
    f32x8 acc[4];
#pragma unroll
    for (int j = 0; j < 4; ++j) acc[j] = zero;

    for (int k0 = 0; k0 < K; k0 += 32) {
        FragBF16 a;
        __builtin_memcpy(&a.u[0], &A[(size_t)arow * K + k0 + half * 16], 32);
#pragma unroll
        for (int j = 0; j < 4; ++j) {
            const size_t tile = (size_t)((n0 >> 4) + j);
            const unsigned short* bp =
                &Bp[((tile * Qn + (k0 >> 5)) * 32 + lane) * 16];
            FragBF16 b;
            __builtin_memcpy(&b.u[0], bp, 32);
            acc[j] = __builtin_amdgcn_wmma_f32_16x16x32_bf16(
                false, a.v, false, b.v, (short)0, acc[j], false, false);
        }
    }

    const int rbase = m0 + half * 8;               // C/D row = v + (lane/16)*8
#pragma unroll
    for (int j = 0; j < 4; ++j) {
        const int bcol = n0 + j * 16 + l16;
        const float bv = bias ? bias[bcol] : 0.0f;
#pragma unroll
        for (int v = 0; v < 8; ++v)
            C[(size_t)(rbase + v) * N + bcol] = acc[j][v] + bv;
    }
}

// ---------------------------------------------------------------------------
// Sequential LSTM recurrence, one layer, both directions (blockIdx.x = dir).
// 1024 threads = 32 waves; each wave owns 8 column tiles (32*8*16 = 4096
// gates). A fragment: all 16 rows broadcast h (rows independent in WMMA,
// row 0 is the live result). W in repacked fragment layout (coalesced b128).
// P[t] (16KB) is DMA'd into LDS by the Tensor Data Mover, overlapped with
// the 8MB WMMA weight stream of the same step.
// ---------------------------------------------------------------------------
__global__ void __launch_bounds__(1024)
lstm_recurrence(const float* __restrict__ Pf, const float* __restrict__ Pb,
                const unsigned short* __restrict__ Wf,
                const unsigned short* __restrict__ Wb,
                float* __restrict__ outs,     // [S][2H] or nullptr
                float* __restrict__ finalH,   // layer base: +dir*H
                float* __restrict__ finalC,
                int S, int writeOuts) {
    constexpr int H  = 1024;
    constexpr int G  = 4 * H;
    constexpr int Qn = H / 32;                 // 32 k-chunks
    __shared__ unsigned short h_sh[H];         // h_{t-1} in bf16
    __shared__ float gates[G];
    __shared__ float p_sh[G];                  // TDM destination for P[t]

    const int dir = blockIdx.x;
    const float* P = dir ? Pb : Pf;
    const unsigned short* W = dir ? Wb : Wf;

    const int tid  = threadIdx.x;
    const int lane = tid & 31;
    const int wave = tid >> 5;                 // 0..31
    const int half = lane >> 4;
    const int l16  = lane & 15;

    h_sh[tid] = 0;
    float c = 0.0f, hreg = 0.0f;
    __syncthreads();

#if HAVE_TDM
    const unsigned ldsP = (unsigned)(uintptr_t)(void*)p_sh;   // LDS byte offset
#endif

    const f32x8 zero = {};
    for (int step = 0; step < S; ++step) {
        const int t = dir ? (S - 1 - step) : step;

#if HAVE_TDM
        if (wave == 0) {
            // D# group0: count=1 | lds_addr | global_addr | type=2
            const unsigned long long ga =
                (unsigned long long)(uintptr_t)(P + (size_t)t * G);
            u32x4 g0 = { 1u, ldsP,
                         (unsigned)(ga & 0xFFFFFFFFu),
                         (unsigned)((ga >> 32) & 0x01FFFFFFu) | (2u << 30) };
            // D# group1: data_size=4B(code 2); tensor_dim0=4096; tensor_dim1=1;
            //            tile_dim0=4096; tile_dim1=1; dim0_stride=4096
            i32x8 g1 = { (int)(2u << 16),
                         (int)((4096u & 0xFFFFu) << 16),
                         (int)(1u << 16),
                         (int)(4096u << 16),
                         1,
                         4096,
                         0,
                         0 };
            i32x4 gz4  = { 0, 0, 0, 0 };
            i32x8 gz8  = { 0, 0, 0, 0, 0, 0, 0, 0 };
            // 6-arg form (clang-23 / therock headers):
            //   (g0, g1, g2, g3, extra, cpol)
            __builtin_amdgcn_tensor_load_to_lds(g0, g1, gz4, gz4, gz8, 0);
        }
#endif

        f32x8 acc[8];
#pragma unroll
        for (int i = 0; i < 8; ++i) acc[i] = zero;

        for (int k0 = 0; k0 < H; k0 += 32) {
            // A fragment: every row carries h (uniform LDS broadcast loads)
            FragBF16 a;
            __builtin_memcpy(&a.u[0], &h_sh[k0 + half * 16], 32);
#pragma unroll
            for (int tl = 0; tl < 8; ++tl) {
                const size_t tile = (size_t)(wave * 8 + tl);
                const unsigned short* bp =
                    &W[((tile * Qn + (k0 >> 5)) * 32 + lane) * 16];
                FragBF16 b;
                __builtin_memcpy(&b.u[0], bp, 32);
                acc[tl] = __builtin_amdgcn_wmma_f32_16x16x32_bf16(
                    false, a.v, false, b.v, (short)0, acc[tl], false, false);
            }
        }

#if HAVE_TDM
        if (wave == 0) __builtin_amdgcn_s_wait_tensorcnt(0);
        __syncthreads();                       // p_sh ready; h_sh reads done
#pragma unroll
        for (int tl = 0; tl < 8; ++tl) {
            const int n = wave * 128 + tl * 16 + l16;
            if (half == 0) gates[n] = acc[tl][0] + p_sh[n];
        }
#else
        __syncthreads();                       // h_sh reads done
#pragma unroll
        for (int tl = 0; tl < 8; ++tl) {
            const int n = wave * 128 + tl * 16 + l16;
            if (half == 0) gates[n] = acc[tl][0] + P[(size_t)t * G + n];
        }
#endif
        __syncthreads();                       // gates ready

        // elementwise cell update: thread tid owns channel tid
        const float iv = gates[tid];
        const float fv = gates[H + tid];
        const float gv = gates[2 * H + tid];
        const float ov = gates[3 * H + tid];
        c    = sigf(fv) * c + sigf(iv) * tanh_fast(gv);
        hreg = sigf(ov) * tanh_fast(c);
        h_sh[tid] = f2bf(hreg);
        if (writeOuts)
            outs[(size_t)t * (2 * H) + dir * H + tid] = hreg;
        __syncthreads();                       // h_sh (and p_sh reuse) safe
    }
    finalH[dir * H + tid] = hreg;
    finalC[dir * H + tid] = c;
}

// ---------------------------------------------------------------------------
// Host-side orchestration
// ---------------------------------------------------------------------------
extern "C" void kernel_launch(void* const* d_in, const int* in_sizes, int n_in,
                              void* d_out, int out_size, void* d_ws, size_t ws_size,
                              hipStream_t stream) {
    constexpr int S = 512, D = 1024, H = 1024;
    constexpr int G  = 4 * H;         // 4096
    constexpr int K1 = 2 * H;         // 2048

    const float* x      = (const float*)d_in[0];
    const float* w_ih0f = (const float*)d_in[1];
    const float* w_hh0f = (const float*)d_in[2];
    const float* b0f    = (const float*)d_in[3];
    const float* w_ih0b = (const float*)d_in[4];
    const float* w_hh0b = (const float*)d_in[5];
    const float* b0b    = (const float*)d_in[6];
    const float* w_ih1f = (const float*)d_in[7];
    const float* w_hh1f = (const float*)d_in[8];
    const float* b1f    = (const float*)d_in[9];
    const float* w_ih1b = (const float*)d_in[10];
    const float* w_hh1b = (const float*)d_in[11];
    const float* b1b    = (const float*)d_in[12];

    float* out    = (float*)d_out;
    float* finalH = out;              // [2][2][H] = 4096
    float* finalC = out + 4096;       // 4096
    float* outs0  = out + 8192;       // [S][2H] = 1048576

    // --- workspace carve-out ---
    char* ws = (char*)d_ws;
    auto take_us = [&](size_t n) {
        unsigned short* p = (unsigned short*)ws;
        ws += n * sizeof(unsigned short);
        return p;
    };
    unsigned short* xbf    = take_us((size_t)S * D);
    unsigned short* o0bf   = take_us((size_t)S * K1);
    unsigned short* ih0f_p = take_us((size_t)G * D);    // repacked weights
    unsigned short* ih0b_p = take_us((size_t)G * D);
    unsigned short* hh0f_p = take_us((size_t)G * H);
    unsigned short* hh0b_p = take_us((size_t)G * H);
    unsigned short* ih1f_p = take_us((size_t)G * K1);
    unsigned short* ih1b_p = take_us((size_t)G * K1);
    unsigned short* hh1f_p = take_us((size_t)G * H);
    unsigned short* hh1b_p = take_us((size_t)G * H);
    ws = (char*)(((uintptr_t)ws + 255) & ~(uintptr_t)255);
    auto take_f = [&](size_t n) {
        float* p = (float*)ws;
        ws += n * sizeof(float);
        return p;
    };
    float* P0f = take_f((size_t)S * G);
    float* P0b = take_f((size_t)S * G);
    float* P1f = take_f((size_t)S * G);
    float* P1b = take_f((size_t)S * G);

    auto cvt = [&](const float* src, unsigned short* dst, int n) {
        cvt_f32_bf16<<<(n + 255) / 256, 256, 0, stream>>>(src, dst, n);
    };
    auto repack = [&](const float* src, unsigned short* dst, int N, int K) {
        const int total = (N / 16) * (K / 32) * 32;
        repack_w_bf16<<<(total + 255) / 256, 256, 0, stream>>>(src, dst, N, K);
    };

    // 1) operand conversions / weight repack (fragment-contiguous bf16)
    cvt(x, xbf, S * D);
    repack(w_ih0f, ih0f_p, G, D);
    repack(w_ih0b, ih0b_p, G, D);
    repack(w_hh0f, hh0f_p, G, H);
    repack(w_hh0b, hh0b_p, G, H);
    repack(w_ih1f, ih1f_p, G, K1);
    repack(w_ih1b, ih1b_p, G, K1);
    repack(w_hh1f, hh1f_p, G, H);
    repack(w_hh1b, hh1b_p, G, H);

    // 2) layer-0 input projections: P0 = x @ W_ih0^T + b0   [S x 4H]
    {
        const int jobs   = (S / 16) * (G / 64);   // 2048 wave-jobs
        const int blocks = jobs / 8;
        gemm_bf16_wmma<<<blocks, 256, 0, stream>>>(xbf, ih0f_p, b0f, P0f, S, G, D);
        gemm_bf16_wmma<<<blocks, 256, 0, stream>>>(xbf, ih0b_p, b0b, P0b, S, G, D);
    }

    // 3) layer-0 recurrence (fwd + bwd concurrently), writes outs0 + finals
    lstm_recurrence<<<2, 1024, 0, stream>>>(P0f, P0b, hh0f_p, hh0b_p,
                                            outs0, finalH, finalC, S, 1);

    // 4) outs0 -> bf16 for layer-1 GEMM input
    cvt(outs0, o0bf, S * K1);

    // 5) layer-1 input projections: P1 = outs0 @ W_ih1^T + b1  (K = 2048)
    {
        const int jobs   = (S / 16) * (G / 64);
        const int blocks = jobs / 8;
        gemm_bf16_wmma<<<blocks, 256, 0, stream>>>(o0bf, ih1f_p, b1f, P1f, S, G, K1);
        gemm_bf16_wmma<<<blocks, 256, 0, stream>>>(o0bf, ih1b_p, b1b, P1b, S, G, K1);
    }

    // 6) layer-1 recurrence: final states only (slots 2,3 of h/c)
    lstm_recurrence<<<2, 1024, 0, stream>>>(P1f, P1b, hh1f_p, hh1b_p,
                                            nullptr, finalH + 2 * H, finalC + 2 * H,
                                            S, 0);
}
